// GAT_89781996355921
// MI455X (gfx1250) — matrix operand order repaired and checked by reading, not compiled, not run on hardware
//
#include <hip/hip_runtime.h>
#include <hip/hip_bf16.h>
#include <math.h>

// ---------------------------------------------------------------------------
// GAT (3-layer, PyG GATConv semantics) for gfx1250.
// GEMMs for layers 0/1: v_wmma_f32_16x16x32_f16 with B staged through LDS by
// the Tensor Data Mover (tensor_load_to_lds, double-buffered, s_wait_tensorcnt).
// Edge softmax/aggregation via global float atomics (L2-resident on MI455X).
// ---------------------------------------------------------------------------

typedef _Float16 v16h __attribute__((ext_vector_type(16)));
typedef _Float16 v8h  __attribute__((ext_vector_type(8)));
typedef float    v8f  __attribute__((ext_vector_type(8)));
typedef unsigned int u32x4 __attribute__((ext_vector_type(4)));
typedef int          i32x8 __attribute__((ext_vector_type(8)));
typedef int          i32x4 __attribute__((ext_vector_type(4)));

#define NEG_SLOPE 0.2f

// ------------------------- small utility kernels ---------------------------

__global__ void fill_u32(unsigned* __restrict__ p, int n, unsigned v) {
    int i = blockIdx.x * blockDim.x + threadIdx.x;
    if (i < n) p[i] = v;
}

__global__ void cvt_f32_f16(const float* __restrict__ in, _Float16* __restrict__ out, int n) {
    int i = blockIdx.x * blockDim.x + threadIdx.x;
    if (i < n) out[i] = (_Float16)in[i];
}

// out_f16 = f16(tanh(in + bias[col]))  (bias applied post-aggregation, per ref)
__global__ void tanh_bias_cvt(const float* __restrict__ in, const float* __restrict__ bias,
                              _Float16* __restrict__ out, int n, int cols) {
    int i = blockIdx.x * blockDim.x + threadIdx.x;
    if (i < n) {
        int c = i % cols;
        out[i] = (_Float16)tanhf(in[i] + bias[c]);
    }
}

// -------------------- weight pre-pack into B-fragment layout ----------------
// Bpack[((kt*NT + nt)*32 + lane)*16 + i] = W[kt*32 + (lane>>4)*16 + i][nt*16 + (lane&15)]
__global__ void pack_W(const float* __restrict__ W, _Float16* __restrict__ Bp, int K, int Nc) {
    int idx = blockIdx.x * blockDim.x + threadIdx.x;
    int NT = Nc >> 4, KT = K >> 5;
    int total = KT * NT * 32;
    if (idx >= total) return;
    int lane = idx & 31;
    int nt = (idx >> 5) % NT;
    int kt = (idx >> 5) / NT;
    int n  = (nt << 4) + (lane & 15);
    int kb = (kt << 5) + ((lane >> 4) << 4);
    _Float16* dst = Bp + (size_t)idx * 16;
#pragma unroll
    for (int i = 0; i < 16; ++i)
        dst[i] = (_Float16)W[(size_t)(kb + i) * Nc + n];
}

// --------------------- TDM: contiguous 1D copy to LDS ----------------------
// D# per CDNA5 ISA ch.8: group0 = {count=1, lds_addr, global_addr, type=2},
// group1 = {data_size=8B, tensor_dim0=tile_dim0=n8, dim1=1, stride0=n8}.
// Toolchain here uses the 6-arg builtin (g0, g1, g2, g3, g4, cpol).
__device__ __forceinline__ void tdm_load_slab(const _Float16* gsrc, unsigned lds_off,
                                              unsigned bytes) {
    unsigned long long ga = (unsigned long long)(size_t)gsrc;
    unsigned n8 = bytes >> 3;                       // 8-byte elements (<= 65535)
    u32x4 g0;
    g0[0] = 1u;                                     // count=1 valid user descriptor
    g0[1] = lds_off;                                // lds_addr (bytes)
    g0[2] = (unsigned)ga;                           // global_addr[31:0]
    g0[3] = ((unsigned)((ga >> 32) & 0x01FFFFFFull)) | 0x80000000u; // addr[56:32] | type=2
    i32x8 g1;
    g1[0] = 3 << 16;                                // data_size = 3 (8B)
    g1[1] = (int)((n8 & 0xFFFFu) << 16);            // tensor_dim0[15:0]
    g1[2] = (int)(((n8 >> 16) & 0xFFFFu) | (1u << 16)); // dim0[31:16] | tensor_dim1=1
    g1[3] = (int)((n8 & 0xFFFFu) << 16);            // tile_dim0
    g1[4] = 1;                                      // tile_dim1 = 1
    g1[5] = (int)n8;                                // tensor_dim0_stride[31:0]
    g1[6] = 0;
    g1[7] = 0;
    i32x4 z4 = {0, 0, 0, 0};                        // groups 2/3 unused (<=2D)
    i32x8 z8 = {0, 0, 0, 0, 0, 0, 0, 0};
    __builtin_amdgcn_tensor_load_to_lds(g0, g1, z4, z4, z8, 0);
}

// ------------------------------ WMMA GEMM ----------------------------------
// C[M,Nc] = A[M,K] @ B[K,Nc]; A row-major f16, B pre-packed f16, C f32.
// Block = 8 waves covering 2 m-tiles x 16 n-tiles; the 16KB B slab for each
// k-step is TDM-copied into LDS (double-buffered) and shared by all 8 waves.
__global__ __launch_bounds__(256)
void gemm_wmma_tdm(const _Float16* __restrict__ A, const _Float16* __restrict__ Bp,
                   float* __restrict__ C, int M, int K, int Nc) {
    const int NT   = Nc >> 4;                 // 16 n-tiles
    const int KT   = K  >> 5;                 // k-steps of 32
    const int SLAB = NT * 32 * 16;            // halves per k-step slab (8192 = 16KB)
    __shared__ _Float16 smem[2][16 * 32 * 16];

    int Mtiles = M >> 4;
    int wid  = threadIdx.x >> 5;              // 0..7
    int lane = threadIdx.x & 31;
    int ng   = wid & 3;                       // 4 n-tile groups of 4
    int mt   = blockIdx.x * 2 + (wid >> 2);
    if (mt >= Mtiles) mt = Mtiles - 1;        // clamp: duplicate tile, keeps barriers uniform
    int m0 = mt << 4;
    int m  = m0 + (lane & 15);
    int koff = (lane >> 4) << 4;              // 0 or 16

    unsigned lds_base = (unsigned)(size_t)(&smem[0][0]);

    if (wid == 0)                             // prologue: slab for kt=0 into buf 0
        tdm_load_slab(Bp, lds_base, (unsigned)SLAB * 2u);

    v8f acc[4] = {v8f{}, v8f{}, v8f{}, v8f{}};

    for (int kt = 0; kt < KT; ++kt) {
        int buf = kt & 1;
        __syncthreads();                      // everyone done reading smem[buf^1]
        if (wid == 0) {
            if (kt + 1 < KT) {
                tdm_load_slab(Bp + (size_t)(kt + 1) * SLAB,
                              lds_base + (unsigned)((buf ^ 1) * SLAB * 2),
                              (unsigned)SLAB * 2u);
                __builtin_amdgcn_s_wait_tensorcnt(1);   // slab kt landed (in-order)
            } else {
                __builtin_amdgcn_s_wait_tensorcnt(0);
            }
        }
        __syncthreads();                      // smem[buf] ready for all waves

        const _Float16* ap = A + (size_t)m * K + (kt << 5) + koff;
        v8h lo = *(const v8h*)(ap);
        v8h hi = *(const v8h*)(ap + 8);
        v16h a;
#pragma unroll
        for (int i = 0; i < 8; ++i) { a[i] = lo[i]; a[8 + i] = hi[i]; }

        const _Float16* bb = &smem[buf][(((ng << 2) * 32) + lane) << 4];
#pragma unroll
        for (int j = 0; j < 4; ++j) {
            v16h b = *(const v16h*)(bb + (size_t)j * 32 * 16);
            acc[j] = __builtin_amdgcn_wmma_f32_16x16x32_f16(
                false, a, false, b, (short)0, acc[j], false, false);
        }
    }

    int rh = lane >> 4;                       // row half
    int cn = lane & 15;
#pragma unroll
    for (int j = 0; j < 4; ++j) {
        int n0 = ((ng << 2) + j) << 4;
#pragma unroll
        for (int r = 0; r < 8; ++r) {
            int row = m0 + r + (rh << 3);
            C[(size_t)row * Nc + n0 + cn] = acc[j][r];
        }
    }
}

// ---------------------- attention scores per node --------------------------
// asrc[n,h] = sum_c h[n,h*C+c]*a_src[h,c]  (a_* stored [H,C] flat)
__global__ void node_scores(const float* __restrict__ h,
                            const float* __restrict__ a_src, const float* __restrict__ a_dst,
                            float* __restrict__ asrc, float* __restrict__ adst,
                            int N, int H, int Cc) {
    int i = blockIdx.x * blockDim.x + threadIdx.x;   // (n, head)
    if (i >= N * H) return;
    int n = i / H, hh = i % H;
    const float* hp = h + (size_t)n * H * Cc + hh * Cc;
    const float* as = a_src + hh * Cc;
    const float* ad = a_dst + hh * Cc;
    float s0 = 0.f, s1 = 0.f;
    for (int c = 0; c < Cc; ++c) { float v = hp[c]; s0 += v * as[c]; s1 += v * ad[c]; }
    asrc[i] = s0;
    adst[i] = s1;
}

// ----------------------- ordered-float encoding ----------------------------
__device__ __forceinline__ unsigned fenc(float f) {
    unsigned u = __float_as_uint(f);
    return (u & 0x80000000u) ? ~u : (u | 0x80000000u);
}
__device__ __forceinline__ float fdec(unsigned u) {
    return __uint_as_float((u & 0x80000000u) ? (u ^ 0x80000000u) : ~u);
}
__device__ __forceinline__ void edge_sd(const int* ei, int e, int E, int* s, int* d) {
    if (e < E) { *s = ei[e]; *d = ei[E + e]; }
    else       { *s = *d = e - E; }       // appended self loops
}
__device__ __forceinline__ float lrelu(float a) { return a > 0.f ? a : NEG_SLOPE * a; }

// -------------------------- edge softmax passes ----------------------------

__global__ void edge_max(const int* __restrict__ ei, int E, int N, int H,
                         const float* __restrict__ asrc, const float* __restrict__ adst,
                         unsigned* __restrict__ amax) {
    int e = blockIdx.x * blockDim.x + threadIdx.x;
    if (e >= E + N) return;
    int s, d; edge_sd(ei, e, E, &s, &d);
    for (int hh = 0; hh < H; ++hh) {
        float al = lrelu(asrc[s * H + hh] + adst[d * H + hh]);
        atomicMax(&amax[d * H + hh], fenc(al));
    }
}

__global__ void edge_sum(const int* __restrict__ ei, int E, int N, int H,
                         const float* __restrict__ asrc, const float* __restrict__ adst,
                         const unsigned* __restrict__ amax, float* __restrict__ denom) {
    int e = blockIdx.x * blockDim.x + threadIdx.x;
    if (e >= E + N) return;
    int s, d; edge_sd(ei, e, E, &s, &d);
    for (int hh = 0; hh < H; ++hh) {
        float al = lrelu(asrc[s * H + hh] + adst[d * H + hh]);
        atomicAdd(&denom[d * H + hh], __expf(al - fdec(amax[d * H + hh])));
    }
}

// one thread per (edge, head): scatter-add coef * h[src] into hout[dst]
__global__ void edge_msg(const int* __restrict__ ei, int E, int N, int H, int Cc,
                         const float* __restrict__ asrc, const float* __restrict__ adst,
                         const unsigned* __restrict__ amax, const float* __restrict__ denom,
                         const float* __restrict__ h, float* __restrict__ hout) {
    int i = blockIdx.x * blockDim.x + threadIdx.x;
    int ET = E + N;
    if (i >= ET * H) return;
    int e = i / H, hh = i % H;
    int s, d; edge_sd(ei, e, E, &s, &d);
    float al   = lrelu(asrc[s * H + hh] + adst[d * H + hh]);
    float coef = __expf(al - fdec(amax[d * H + hh])) / denom[d * H + hh];
    const float* hp = h    + (size_t)s * H * Cc + hh * Cc;
    float*       op = hout + (size_t)d * H * Cc + hh * Cc;
    for (int c = 0; c < Cc; ++c)
        atomicAdd(&op[c], hp[c] * coef);
}

// ----------------------------- layer 2 (256 -> 2) --------------------------

__global__ void l2_gemm_scores(const _Float16* __restrict__ xh, const float* __restrict__ W2,
                               const float* __restrict__ a_src, const float* __restrict__ a_dst,
                               float* __restrict__ h2, float* __restrict__ asrc,
                               float* __restrict__ adst, int N, int K) {
    int n = blockIdx.x * blockDim.x + threadIdx.x;
    if (n >= N) return;
    const _Float16* xp = xh + (size_t)n * K;
    float a0 = 0.f, a1 = 0.f;
    for (int k = 0; k < K; ++k) {
        float v = (float)xp[k];
        a0 += v * W2[k * 2];
        a1 += v * W2[k * 2 + 1];
    }
    h2[n * 2] = a0; h2[n * 2 + 1] = a1;
    asrc[n] = a0 * a_src[0] + a1 * a_src[1];
    adst[n] = a0 * a_dst[0] + a1 * a_dst[1];
}

__global__ void finalize(const float* __restrict__ hout2, const float* __restrict__ b2,
                         float* __restrict__ out, int N) {
    int n = blockIdx.x * blockDim.x + threadIdx.x;
    if (n >= N) return;
    float l0 = hout2[n * 2] + b2[0];
    float l1 = hout2[n * 2 + 1] + b2[1];
    float m  = fmaxf(l0, l1);
    float e0 = __expf(l0 - m), e1 = __expf(l1 - m);
    float inv = 1.f / (e0 + e1);
    out[n * 2]             = e0 * inv;     // softmax
    out[n * 2 + 1]         = e1 * inv;
    out[2 * N + n * 2]     = l0;           // logits
    out[2 * N + n * 2 + 1] = l1;
}

// ------------------------------- host side ---------------------------------

extern "C" void kernel_launch(void* const* d_in, const int* in_sizes, int n_in,
                              void* d_out, int out_size, void* d_ws, size_t ws_size,
                              hipStream_t stream) {
    const float* x  = (const float*)d_in[0];
    const int*   ei = (const int*)d_in[1];
    const float* W0 = (const float*)d_in[2];
    const float* as0 = (const float*)d_in[3];
    const float* ad0 = (const float*)d_in[4];
    const float* b0 = (const float*)d_in[5];
    const float* W1 = (const float*)d_in[6];
    const float* as1 = (const float*)d_in[7];
    const float* ad1 = (const float*)d_in[8];
    const float* b1 = (const float*)d_in[9];
    const float* W2 = (const float*)d_in[10];
    const float* as2w = (const float*)d_in[11];
    const float* ad2w = (const float*)d_in[12];
    const float* b2 = (const float*)d_in[13];

    const int DIN = 128, H = 4, Cc = 64, HC = 256;
    const int N = in_sizes[0] / DIN;
    const int E = in_sizes[1] / 2;
    const int ET = E + N;

    // workspace carve-out (256B aligned)
    char* ws = (char*)d_ws;
    size_t off = 0;
    auto alloc = [&](size_t bytes) -> void* {
        void* p = ws + off;
        off = (off + bytes + 255) & ~(size_t)255;
        return p;
    };
    _Float16* xh   = (_Float16*)alloc((size_t)N * HC * 2);           // f16 layer input
    _Float16* Bp   = (_Float16*)alloc((size_t)8 * 16 * 32 * 16 * 2); // packed W (max K=256)
    float*    hbuf = (float*)alloc((size_t)N * HC * 4);              // h = x@W
    float*    hout = (float*)alloc((size_t)N * HC * 4);              // aggregated output
    float*    asrc = (float*)alloc((size_t)N * H * 4);
    float*    adst = (float*)alloc((size_t)N * H * 4);
    unsigned* amax = (unsigned*)alloc((size_t)N * H * 4);
    float*    den  = (float*)alloc((size_t)N * H * 4);
    float*    h2   = (float*)alloc((size_t)N * 2 * 4);
    float*    asc2 = (float*)alloc((size_t)N * 4);
    float*    adc2 = (float*)alloc((size_t)N * 4);
    float*    ho2  = (float*)alloc((size_t)N * 2 * 4);
    (void)ws_size; (void)n_in; (void)out_size;

    const int TB = 256;
    auto blk = [](long long n, int t) { return (unsigned)((n + t - 1) / t); };

    // x -> f16
    cvt_f32_f16<<<blk((long long)N * DIN, TB), TB, 0, stream>>>(x, xh, N * DIN);

    struct Lp { const float *W, *as, *ad, *b; int K; };
    Lp L[2] = { {W0, as0, ad0, b0, DIN}, {W1, as1, ad1, b1, HC} };

    for (int l = 0; l < 2; ++l) {
        int K = L[l].K;
        int KT = K / 32, NT = HC / 16;
        pack_W<<<blk(KT * NT * 32, TB), TB, 0, stream>>>(L[l].W, Bp, K, HC);

        int Mtiles = N / 16;
        unsigned gblocks = (unsigned)((Mtiles + 1) / 2);     // 2 m-tiles per block
        gemm_wmma_tdm<<<gblocks, TB, 0, stream>>>(xh, Bp, hbuf, N, K, HC);

        fill_u32<<<blk((long long)N * H, TB), TB, 0, stream>>>(amax, N * H, 0u);
        fill_u32<<<blk((long long)N * H, TB), TB, 0, stream>>>((unsigned*)den, N * H, 0u);
        fill_u32<<<blk((long long)N * HC, TB), TB, 0, stream>>>((unsigned*)hout, N * HC, 0u);

        node_scores<<<blk((long long)N * H, TB), TB, 0, stream>>>(hbuf, L[l].as, L[l].ad,
                                                                  asrc, adst, N, H, Cc);
        edge_max<<<blk(ET, TB), TB, 0, stream>>>(ei, E, N, H, asrc, adst, amax);
        edge_sum<<<blk(ET, TB), TB, 0, stream>>>(ei, E, N, H, asrc, adst, amax, den);
        edge_msg<<<blk((long long)ET * H, TB), TB, 0, stream>>>(ei, E, N, H, Cc, asrc, adst,
                                                                amax, den, hbuf, hout);
        // next layer's f16 input: tanh(hout + b)
        tanh_bias_cvt<<<blk((long long)N * HC, TB), TB, 0, stream>>>(hout, L[l].b, xh,
                                                                     N * HC, HC);
    }

    // layer 2: 256 -> 2, single head
    l2_gemm_scores<<<blk(N, TB), TB, 0, stream>>>(xh, W2, as2w, ad2w, h2, asc2, adc2, N, HC);
    fill_u32<<<blk(N, TB), TB, 0, stream>>>(amax, N, 0u);
    fill_u32<<<blk(N, TB), TB, 0, stream>>>((unsigned*)den, N, 0u);
    fill_u32<<<blk((long long)N * 2, TB), TB, 0, stream>>>((unsigned*)ho2, N * 2, 0u);
    edge_max<<<blk(ET, TB), TB, 0, stream>>>(ei, E, N, 1, asc2, adc2, amax);
    edge_sum<<<blk(ET, TB), TB, 0, stream>>>(ei, E, N, 1, asc2, adc2, amax, den);
    edge_msg<<<blk(ET, TB), TB, 0, stream>>>(ei, E, N, 1, 2, asc2, adc2, amax, den, h2, ho2);
    finalize<<<blk(N, TB), TB, 0, stream>>>(ho2, b2, (float*)d_out, N);
}
